// TeacherNet_81896436400804
// MI455X (gfx1250) — matrix operand order repaired and checked by reading, not compiled, not run on hardware
//
#include <hip/hip_runtime.h>
#include <hip/hip_bf16.h>

// ---------------------------------------------------------------------------
// DGCNN forward (N=16384, K=20) for MI455X (gfx1250, wave32, WMMA).
// All GEMM-shaped work (x@x.T distance tiles, edge MLPs, final linear) runs on
// v_wmma_f32_16x16x32_bf16 with f32 accumulation. The N x N distance matrix is
// never materialized: 64x64 tiles in LDS, fused register-resident top-20, and
// the next candidate tile is staged to LDS asynchronously (ASYNCcnt) while the
// scalar top-K phase runs.
// ---------------------------------------------------------------------------

typedef __attribute__((ext_vector_type(16))) __bf16 v16bf;
typedef __attribute__((ext_vector_type(8)))  __bf16 v8bf;
typedef __attribute__((ext_vector_type(8)))  float  v8f;
typedef __attribute__((ext_vector_type(4)))  int    v4i;

#define NPTS 16384
#define KNN  20

#ifndef __has_builtin
#define __has_builtin(x) 0
#endif
#if defined(__gfx1250__) && __has_builtin(__builtin_amdgcn_global_load_async_to_lds_b128)
#define USE_ASYNC_LDS 1
#else
#define USE_ASYNC_LDS 0
#endif

#if USE_ASYNC_LDS
// Builtin signature (from hipcc diagnostic): pointers to int4 in AS1 / AS3,
// plus constant-int offset and cpol.
typedef __attribute__((address_space(1))) v4i gv4i;
typedef __attribute__((address_space(3))) v4i lv4i;
// 16-byte global -> LDS async copy (per-lane addresses), tracked by ASYNCcnt.
__device__ __forceinline__ void async_copy16(const __bf16* g, __bf16* l) {
  __builtin_amdgcn_global_load_async_to_lds_b128((gv4i*)g, (lv4i*)l, 0, 0);
}
__device__ __forceinline__ void wait_async0() {
#if __has_builtin(__builtin_amdgcn_s_wait_asynccnt)
  __builtin_amdgcn_s_wait_asynccnt(0);
#else
  asm volatile("s_wait_asynccnt 0x0" ::: "memory");
#endif
}
#endif

// A-matrix 16x32 bf16 fragment (ISA 7.12.2): lane m = lane&15, half = lane>>4;
// per lane: contiguous K [half*8, half*8+8) then [16+half*8, 16+half*8+8).
__device__ __forceinline__ v16bf load_afrag(const __bf16* row, int half) {
  union { v16bf v; v8bf h[2]; } u;
  u.h[0] = *reinterpret_cast<const v8bf*>(row + half * 8);
  u.h[1] = *reinterpret_cast<const v8bf*>(row + 16 + half * 8);
  return u.v;
}
// B-matrix 32x16 bf16 fragment: lane n = lane&15, half = lane>>4;
// per lane: K [half*8, half*8+8) then [16+half*8, 16+half*8+8). `row` points at B^T row n.
__device__ __forceinline__ v16bf load_bfrag(const __bf16* row, int half) {
  union { v16bf v; v8bf h[2]; } u;
  u.h[0] = *reinterpret_cast<const v8bf*>(row + half * 8);
  u.h[1] = *reinterpret_cast<const v8bf*>(row + 16 + half * 8);
  return u.v;
}

// ---------------------------------------------------------------------------
// mlp1: x1 = relu(pos @ W1 + b1)   (3 -> 64), plus bf16 copy and |x|^2.
// ---------------------------------------------------------------------------
__global__ void mlp1_kernel(const float* __restrict__ pos, const float* __restrict__ W1,
                            const float* __restrict__ b1, float* __restrict__ x1f,
                            __bf16* __restrict__ x1b, float* __restrict__ sqn1) {
  int p = blockIdx.x * blockDim.x + threadIdx.x;
  if (p >= NPTS) return;
  float p0 = pos[p * 3 + 0], p1 = pos[p * 3 + 1], p2 = pos[p * 3 + 2];
  float s = 0.f;
#pragma unroll 8
  for (int o = 0; o < 64; ++o) {
    float v = fmaf(p0, W1[o], fmaf(p1, W1[64 + o], fmaf(p2, W1[128 + o], b1[o])));
    v = fmaxf(v, 0.f);
    x1f[(size_t)p * 64 + o] = v;
    x1b[(size_t)p * 64 + o] = (__bf16)v;
    s = fmaf(v, v, s);
  }
  sqn1[p] = s;
}

// ---------------------------------------------------------------------------
// Transpose + convert weights: W[kin][kout] f32 -> Wt[kout][kin] bf16 so that
// WMMA B-fragments are contiguous loads (weights then live in L2).
// ---------------------------------------------------------------------------
__global__ void wtrans_kernel(const float* __restrict__ W, __bf16* __restrict__ Wt,
                              int kin, int kout) {
  int t = blockIdx.x * blockDim.x + threadIdx.x;
  if (t >= kin * kout) return;
  int k = t / kout, n = t % kout;
  Wt[(size_t)n * kin + k] = (__bf16)W[(size_t)k * kout + n];
}

// ---------------------------------------------------------------------------
// Convert features to bf16 and compute squared norms (for the next kNN).
// ---------------------------------------------------------------------------
__global__ void tobf_sqn_kernel(const float* __restrict__ X, __bf16* __restrict__ Xb,
                                float* __restrict__ sqn, int F) {
  int p = blockIdx.x * blockDim.x + threadIdx.x;
  if (p >= NPTS) return;
  float s = 0.f;
  for (int f = 0; f < F; ++f) {
    float v = X[(size_t)p * F + f];
    Xb[(size_t)p * F + f] = (__bf16)v;
    s = fmaf(v, v, s);
  }
  if (sqn) sqn[p] = s;
}

// ---------------------------------------------------------------------------
// Fused kNN: block owns 64 query rows; loop over 64-candidate tiles.
// Pipeline per tile:  barrier -> WMMA(S) -> barrier -> issue async stage of
// tile ct+1 -> register top-K on tile ct -> s_wait_asynccnt.
// score = |xc|^2 - 2<xq,xc>  (|xq|^2 is row-constant, cannot change ranking).
// ---------------------------------------------------------------------------
template <int F>
__global__ void __launch_bounds__(64)
knn_kernel(const __bf16* __restrict__ Xb, const float* __restrict__ sqn,
           int* __restrict__ knnidx) {
  constexpr int RS = F + 8;                 // padded bf16 row stride (16B aligned)
  __shared__ __bf16 Xq[64 * RS];
  __shared__ __bf16 Xc[64 * RS];
  __shared__ float  S[64 * 68];
  __shared__ float  sqc[2][64];             // double-buffered norms

  const int t = threadIdx.x;
  const int lane = t & 31;
  const int half = lane >> 4;
  const int wave = t >> 5;                  // 0..1
  const int row0 = blockIdx.x * 64;

  // stage this block's 64 query rows + candidate tile 0
#if USE_ASYNC_LDS
  {
    const __bf16* qs = Xb + (size_t)(row0 + t) * F;
    const __bf16* cs = Xb + (size_t)t * F;
#pragma unroll
    for (int i = 0; i < F / 8; ++i) {
      async_copy16(qs + i * 8, &Xq[t * RS + i * 8]);
      async_copy16(cs + i * 8, &Xc[t * RS + i * 8]);
    }
    sqc[0][t] = sqn[t];
    wait_async0();
  }
#else
  {
    const uint4* qs = reinterpret_cast<const uint4*>(Xb + (size_t)(row0 + t) * F);
    const uint4* cs = reinterpret_cast<const uint4*>(Xb + (size_t)t * F);
    uint4* qd = reinterpret_cast<uint4*>(&Xq[t * RS]);
    uint4* cd = reinterpret_cast<uint4*>(&Xc[t * RS]);
#pragma unroll
    for (int i = 0; i < F / 8; ++i) { qd[i] = qs[i]; cd[i] = cs[i]; }
    sqc[0][t] = sqn[t];
  }
#endif

  float bestd[KNN];
  int   besti[KNN];
#pragma unroll
  for (int q = 0; q < KNN; ++q) { bestd[q] = 3.0e38f; besti[q] = 0; }

  const int ntiles = NPTS / 64;
  for (int ct = 0; ct < ntiles; ++ct) {
    const int buf = ct & 1;
    __syncthreads();                        // Xc tile ct + sqc[buf] visible

    // 64x64 S tile = 16 WMMA tiles; 8 per wave.
    for (int tile = wave; tile < 16; tile += 2) {
      const int tm = tile >> 2, tn = tile & 3;
      v8f c = {};
#pragma unroll
      for (int kk = 0; kk < F / 32; ++kk) {
        v16bf a = load_afrag(&Xq[(tm * 16 + (lane & 15)) * RS + kk * 32], half);
        v16bf b = load_bfrag(&Xc[(tn * 16 + (lane & 15)) * RS + kk * 32], half);
        c = __builtin_amdgcn_wmma_f32_16x16x32_bf16(false, a, false, b,
                                                    (short)0, c, false, false);
      }
#pragma unroll
      for (int v = 0; v < 8; ++v)
        S[(tm * 16 + v + half * 8) * 68 + tn * 16 + (lane & 15)] = c[v];
    }
    __syncthreads();                        // S ready; Xc free to overwrite

    // issue staging of tile ct+1 now; it overlaps the scalar top-K below
    // (top-K reads only S and sqc[buf], never Xc).
    if (ct + 1 < ntiles) {
#if USE_ASYNC_LDS
      const __bf16* cs = Xb + (size_t)((ct + 1) * 64 + t) * F;
#pragma unroll
      for (int i = 0; i < F / 8; ++i)
        async_copy16(cs + i * 8, &Xc[t * RS + i * 8]);
#else
      const uint4* cs = reinterpret_cast<const uint4*>(Xb + (size_t)((ct + 1) * 64 + t) * F);
      uint4* cd = reinterpret_cast<uint4*>(&Xc[t * RS]);
#pragma unroll
      for (int i = 0; i < F / 8; ++i) cd[i] = cs[i];
#endif
      sqc[buf ^ 1][t] = sqn[(ct + 1) * 64 + t];
    }

    // per-row top-K update (thread t owns query row t)
    const int gr = row0 + t;
    for (int j = 0; j < 64; ++j) {
      const int gj = ct * 64 + j;
      float d = fmaf(-2.0f, S[t * 68 + j], sqc[buf][j]);
      if (gj == gr) d = -3.0e38f;           // self always included (d2 == 0)
      if (d < bestd[KNN - 1]) {
        bestd[KNN - 1] = d; besti[KNN - 1] = gj;
#pragma unroll
        for (int q = KNN - 1; q > 0; --q) { // bubble new element into place
          if (bestd[q] < bestd[q - 1]) {
            float td = bestd[q]; bestd[q] = bestd[q - 1]; bestd[q - 1] = td;
            int   ti = besti[q]; besti[q] = besti[q - 1]; besti[q - 1] = ti;
          }
        }
      }
    }

#if USE_ASYNC_LDS
    if (ct + 1 < ntiles) wait_async0();     // our Xc writes land before barrier
#endif
  }
#pragma unroll
  for (int q = 0; q < KNN; ++q) knnidx[(size_t)(row0 + t) * KNN + q] = besti[q];
}

// ---------------------------------------------------------------------------
// DynamicEdgeConv: block owns 4 points x 20 neighbors = 80 rows (5 M-tiles).
// feat = [x_i | x_j - x_i] built bf16 in LDS; H = feat @ We via WMMA in
// 64-column chunks; out = relu(max_k H + b)  (relu/max commute).
// ---------------------------------------------------------------------------
template <int FIN, int FOUT>
__global__ void __launch_bounds__(256)
edgeconv_kernel(const float* __restrict__ X, const int* __restrict__ knnidx,
                const __bf16* __restrict__ Wt,   // [FOUT][2*FIN] bf16 (= We^T)
                const float* __restrict__ bias, float* __restrict__ Y) {
  constexpr int KIN = 2 * FIN;
  constexpr int RS = KIN + 8;
  __shared__ __bf16 feat[80 * RS];
  __shared__ float  H[80 * 68];

  const int t = threadIdx.x;
  const int lane = t & 31;
  const int half = lane >> 4;
  const int wave = t >> 5;                  // 0..7
  const int p0 = blockIdx.x * 4;

  if (t < 80) {                             // one thread builds one edge row
    const int p = t / KNN, k = t % KNN;
    const int gp = p0 + p;
    const int gj = knnidx[(size_t)gp * KNN + k];
    const float* xi = X + (size_t)gp * FIN;
    const float* xj = X + (size_t)gj * FIN;
    __bf16* fr = &feat[t * RS];
#pragma unroll 4
    for (int f = 0; f < FIN; ++f) {
      float a = xi[f];
      fr[f] = (__bf16)a;
      fr[FIN + f] = (__bf16)(xj[f] - a);
    }
  }
  __syncthreads();

  for (int c0 = 0; c0 < FOUT; c0 += 64) {
    // 80 x 64 chunk = 5x4 = 20 WMMA tiles over 8 waves
    for (int tile = wave; tile < 20; tile += 8) {
      const int tm = tile >> 2, tn = tile & 3;
      v8f c = {};
#pragma unroll
      for (int kk = 0; kk < KIN / 32; ++kk) {
        v16bf a = load_afrag(&feat[(tm * 16 + (lane & 15)) * RS + kk * 32], half);
        v16bf b = load_bfrag(Wt + (size_t)(c0 + tn * 16 + (lane & 15)) * KIN + kk * 32, half);
        c = __builtin_amdgcn_wmma_f32_16x16x32_bf16(false, a, false, b,
                                                    (short)0, c, false, false);
      }
#pragma unroll
      for (int v = 0; v < 8; ++v)
        H[(tm * 16 + v + half * 8) * 68 + tn * 16 + (lane & 15)] = c[v];
    }
    __syncthreads();

    { // max over 20 neighbors + bias + relu; 256 threads = 4 pts x 64 cols
      const int p = t >> 6, c = t & 63;
      float m = -3.0e38f;
#pragma unroll
      for (int k = 0; k < KNN; ++k) m = fmaxf(m, H[(p * KNN + k) * 68 + c]);
      Y[(size_t)(p0 + p) * FOUT + c0 + c] = fmaxf(m + bias[c0 + c], 0.f);
    }
    __syncthreads();
  }
}

// ---------------------------------------------------------------------------
// mlp2: out = x3 @ W2 + b2  (16384x256 @ 256x128), plain WMMA GEMM, A and B
// fragments straight from global (rows are 16B-aligned contiguous bf16).
// ---------------------------------------------------------------------------
__global__ void __launch_bounds__(256)
mlp2_kernel(const __bf16* __restrict__ Xb,   // [N][256] bf16
            const __bf16* __restrict__ W2t,  // [128][256] bf16 (= W2^T)
            const float* __restrict__ b2, float* __restrict__ out) {
  const int t = threadIdx.x;
  const int lane = t & 31;
  const int half = lane >> 4;
  const int wave = t >> 5;
  const int row0 = blockIdx.x * 64;
  for (int tile = wave; tile < 32; tile += 8) {   // 4 M-tiles x 8 N-tiles
    const int tm = tile >> 3, tn = tile & 7;
    v8f c = {};
#pragma unroll
    for (int kk = 0; kk < 8; ++kk) {
      v16bf a = load_afrag(Xb + (size_t)(row0 + tm * 16 + (lane & 15)) * 256 + kk * 32, half);
      v16bf b = load_bfrag(W2t + (size_t)(tn * 16 + (lane & 15)) * 256 + kk * 32, half);
      c = __builtin_amdgcn_wmma_f32_16x16x32_bf16(false, a, false, b,
                                                  (short)0, c, false, false);
    }
    const int col = tn * 16 + (lane & 15);
    const float bb = b2[col];
#pragma unroll
    for (int v = 0; v < 8; ++v)
      out[(size_t)(row0 + tm * 16 + v + half * 8) * 128 + col] = c[v] + bb;
  }
}

// ---------------------------------------------------------------------------
extern "C" void kernel_launch(void* const* d_in, const int* in_sizes, int n_in,
                              void* d_out, int out_size, void* d_ws, size_t ws_size,
                              hipStream_t stream) {
  const float* pos = (const float*)d_in[0];
  const float* W1  = (const float*)d_in[1];
  const float* b1  = (const float*)d_in[2];
  const float* We1 = (const float*)d_in[3];
  const float* be1 = (const float*)d_in[4];
  const float* We2 = (const float*)d_in[5];
  const float* be2 = (const float*)d_in[6];
  const float* W2  = (const float*)d_in[7];
  const float* b2  = (const float*)d_in[8];
  float* out = (float*)d_out;

  char* ws = (char*)d_ws;
  size_t off = 0;
  auto alloc = [&](size_t bytes) -> void* {
    off = (off + 255) & ~(size_t)255;
    void* p = ws + off;
    off += bytes;
    return p;
  };
  const size_t N = NPTS;
  float*  x1f  = (float*) alloc(N * 64 * 4);
  __bf16* x1b  = (__bf16*)alloc(N * 64 * 2);
  float*  sqn1 = (float*) alloc(N * 4);
  int*    knn1 = (int*)   alloc(N * KNN * 4);
  float*  x2f  = (float*) alloc(N * 128 * 4);
  __bf16* x2b  = (__bf16*)alloc(N * 128 * 2);
  float*  sqn2 = (float*) alloc(N * 4);
  int*    knn2 = (int*)   alloc(N * KNN * 4);
  float*  x3f  = (float*) alloc(N * 256 * 4);
  __bf16* x3b  = (__bf16*)alloc(N * 256 * 2);
  __bf16* We1t = (__bf16*)alloc(128 * 128 * 2);
  __bf16* We2t = (__bf16*)alloc(256 * 256 * 2);
  __bf16* W2t  = (__bf16*)alloc(128 * 256 * 2);

  mlp1_kernel<<<NPTS / 128, 128, 0, stream>>>(pos, W1, b1, x1f, x1b, sqn1);
  wtrans_kernel<<<(128 * 128) / 256, 256, 0, stream>>>(We1, We1t, 128, 128);
  wtrans_kernel<<<(256 * 256) / 256, 256, 0, stream>>>(We2, We2t, 256, 256);
  wtrans_kernel<<<(256 * 128) / 256, 256, 0, stream>>>(W2, W2t, 256, 128);

  knn_kernel<64><<<NPTS / 64, 64, 0, stream>>>(x1b, sqn1, knn1);
  edgeconv_kernel<64, 128><<<NPTS / 4, 256, 0, stream>>>(x1f, knn1, We1t, be1, x2f);
  tobf_sqn_kernel<<<NPTS / 128, 128, 0, stream>>>(x2f, x2b, sqn2, 128);

  knn_kernel<128><<<NPTS / 64, 64, 0, stream>>>(x2b, sqn2, knn2);
  edgeconv_kernel<128, 256><<<NPTS / 4, 256, 0, stream>>>(x2f, knn2, We2t, be2, x3f);
  tobf_sqn_kernel<<<NPTS / 128, 128, 0, stream>>>(x3f, x3b, nullptr, 256);

  mlp2_kernel<<<NPTS / 64, 256, 0, stream>>>(x3b, W2t, b2, out);

  (void)in_sizes; (void)n_in; (void)out_size; (void)ws_size;
}
